// OuroBlock_46462956208932
// MI455X (gfx1250) — matrix operand order repaired and checked by reading, not compiled, not run on hardware
//
#include <hip/hip_runtime.h>
#include <hip/hip_bf16.h>
#include <math.h>

#define DD 768
#define NH 12
#define HD 64
#define BB 2
#define TT 1024
#define MTOK (BB*TT)   // 2048 tokens

typedef __attribute__((ext_vector_type(16))) _Float16 v16h;
typedef __attribute__((ext_vector_type(8)))  float    v8f;

#define BPITCH 40   // halves per LDS row: 80B, keeps b128 alignment, conflict-free

// ---------------------------------------------------------------------------
// Generic batched GEMM:  C = beta*C + act(alpha * A@B + bias)
//   A: [batch?][M,K] f32 row-major   B: [batch?][K,N] (or [N,K] if transB)
//   Block = 4 waves = 64x64 output tile. Wave w owns rows m0+16w, all 64 cols:
//   4 f32 accumulators, 4 x v_wmma_f32_16x16x32_f16 per 32-wide K step.
//   B panel staged in LDS K-contiguous per column (Bs[n][k]) so each lane's
//   fragment is 2 x ds_load_b128; staging is 2 x ds_store_b128 per thread.
//   All shapes in this model have M%64==0, N%64==0, K%32==0.
// ---------------------------------------------------------------------------
__global__ __launch_bounds__(128) void gemm_f16_wmma(
    const float* __restrict__ A, const float* __restrict__ Bm,
    const float* __restrict__ bias, float* __restrict__ C,
    int M, int N, int K,
    long long sA, long long sB, long long sC,
    int transB, int act, float alpha, float beta)
{
  __shared__ __align__(16) _Float16 Bs[64][BPITCH];   // [n][k], 5.1 KB

  const int lane = threadIdx.x;
  const int wave = threadIdx.y;
  const int tid  = wave * 32 + lane;
  const int n0 = blockIdx.x * 64;
  const int m0 = blockIdx.y * 64;

  const float* Ab = A  + (long long)blockIdx.z * sA;
  const float* Bb = Bm + (long long)blockIdx.z * sB;
  float*       Cb = C  + (long long)blockIdx.z * sC;

  // A fragment addressing (16-bit A 16x32), K-contiguous to match B:
  //  lanes 0-15: M=lane, K-base 0 ; lanes 16-31: M=lane-16, K-base 16
  //  element e -> K = base+e
  const int arow = m0 + wave * 16 + (lane & 15);
  const int akb  = (lane >> 4) << 4;              // 0 or 16
  // B fragment (16-bit B 32x16): col = lane&15; lanes<16 K=0..15, else 16..31
  const int ncol = lane & 15;
  const int bkb  = (lane >> 4) << 4;              // 0 or 16

  // staging map: thread -> (column n, 16-wide k chunk)
  const int sn = tid >> 1;
  const int skg = (tid & 1) << 4;

  v8f acc0 = {}, acc1 = {}, acc2 = {}, acc3 = {};

  for (int k0 = 0; k0 < K; k0 += 32) {
    __syncthreads();                               // prev reads of Bs done
    v16h tmp;
    if (!transB) {
      // gather K-column of B: B[k0+skg+e][n0+sn]
      const float* bp = Bb + (long long)(k0 + skg) * N + (n0 + sn);
      __builtin_prefetch(bp + 32LL * N, 0, 1);     // next K panel
#pragma unroll
      for (int e = 0; e < 16; ++e) tmp[e] = (_Float16)bp[(long long)e * N];
    } else {
      // B is [N,K] row-major: rows are K-contiguous (coalesced b128 loads)
      const float* bp = Bb + (long long)(n0 + sn) * K + (k0 + skg);
      __builtin_prefetch(bp + 32, 0, 1);
#pragma unroll
      for (int e = 0; e < 16; ++e) tmp[e] = (_Float16)bp[e];
    }
    *(v16h*)&Bs[sn][skg] = tmp;                    // 2 x ds_store_b128
    __syncthreads();

    const float* ap = Ab + (long long)arow * K + (k0 + akb);
    __builtin_prefetch(ap + 32, 0, 1);
    v16h af;
#pragma unroll
    for (int e = 0; e < 16; ++e) {
      int vg = e >> 1;
      int koff = 2 * vg + (e & 1);
      af[e] = (_Float16)ap[koff];
    }

    // fragments: 16 K-contiguous halves per lane -> 2 x ds_load_b128 each
    v16h bf0 = *(const v16h*)&Bs[ncol][bkb];
    v16h bf1 = *(const v16h*)&Bs[16 + ncol][bkb];
    v16h bf2 = *(const v16h*)&Bs[32 + ncol][bkb];
    v16h bf3 = *(const v16h*)&Bs[48 + ncol][bkb];

    acc0 = __builtin_amdgcn_wmma_f32_16x16x32_f16(false, af, false, bf0, (short)0, acc0, false, false);
    acc1 = __builtin_amdgcn_wmma_f32_16x16x32_f16(false, af, false, bf1, (short)0, acc1, false, false);
    acc2 = __builtin_amdgcn_wmma_f32_16x16x32_f16(false, af, false, bf2, (short)0, acc2, false, false);
    acc3 = __builtin_amdgcn_wmma_f32_16x16x32_f16(false, af, false, bf3, (short)0, acc3, false, false);
  }

  // C/D layout: VGPR v -> M = v (+8 for lanes 16-31), N = lane&15
  const int rbase = m0 + wave * 16 + ((lane >> 4) << 3);
  v8f accs[4] = {acc0, acc1, acc2, acc3};
#pragma unroll
  for (int j = 0; j < 4; ++j) {
    const int ccol = n0 + (j << 4) + ncol;
    const float bv = bias ? bias[ccol] : 0.0f;
#pragma unroll
    for (int v = 0; v < 8; ++v) {
      long long idx = (long long)(rbase + v) * N + ccol;
      float r = alpha * accs[j][v] + bv;
      if (act == 1)      r = r / (1.0f + __expf(-r));      // silu
      else if (act == 2) r = 1.0f / (1.0f + __expf(-r));   // sigmoid
      float prev = (beta != 0.0f) ? beta * Cb[idx] : 0.0f;
      Cb[idx] = prev + r;
    }
  }
}

// ---------------------------------------------------------------------------
// LayerNorm: one wave per row of width DD
// ---------------------------------------------------------------------------
__global__ void layernorm_kernel(const float* __restrict__ X,
                                 const float* __restrict__ g,
                                 const float* __restrict__ b,
                                 float* __restrict__ Y, int M)
{
  int row = blockIdx.x * blockDim.y + threadIdx.y;
  if (row >= M) return;
  int lane = threadIdx.x;
  const float* x = X + (long long)row * DD;
  float s = 0.f, ss = 0.f;
  for (int j = lane; j < DD; j += 32) { float v = x[j]; s += v; ss += v * v; }
#pragma unroll
  for (int o = 16; o > 0; o >>= 1) {
    s  += __shfl_xor(s,  o, 32);
    ss += __shfl_xor(ss, o, 32);
  }
  float m    = s / (float)DD;
  float var  = ss / (float)DD - m * m;
  float rstd = rsqrtf(var + 1e-5f);
  float* y = Y + (long long)row * DD;
  for (int j = lane; j < DD; j += 32)
    y[j] = (x[j] - m) * rstd * g[j] + b[j];
}

// Row L2 normalize in place: x /= max(||x||, 1e-5)
__global__ void rownorm_kernel(float* __restrict__ X, int M)
{
  int row = blockIdx.x * blockDim.y + threadIdx.y;
  if (row >= M) return;
  int lane = threadIdx.x;
  float* x = X + (long long)row * DD;
  float ss = 0.f;
  for (int j = lane; j < DD; j += 32) { float v = x[j]; ss += v * v; }
#pragma unroll
  for (int o = 16; o > 0; o >>= 1) ss += __shfl_xor(ss, o, 32);
  float inv = 1.0f / fmaxf(sqrtf(ss), 1e-5f);
  for (int j = lane; j < DD; j += 32) x[j] *= inv;
}

// ---------------------------------------------------------------------------
// RoPE applied in-place to q,k slices of qkv [B,T,3*DD]; cos/sin [T,HD]
// one thread per (b,t,h,d0) with d0 in [0,32)
// ---------------------------------------------------------------------------
__global__ void rope_kernel(float* __restrict__ qkv,
                            const float* __restrict__ cosT,
                            const float* __restrict__ sinT)
{
  long long tid = blockIdx.x * (long long)blockDim.x + threadIdx.x;
  long long total = (long long)BB * TT * NH * (HD / 2);
  if (tid >= total) return;
  int d0 = (int)(tid & 31);
  long long r = tid >> 5;
  int h = (int)(r % NH); r /= NH;
  int t = (int)(r % TT); int b = (int)(r / TT);
  float c0 = cosT[t * HD + d0],      s0 = sinT[t * HD + d0];
  float c1 = cosT[t * HD + d0 + 32], s1 = sinT[t * HD + d0 + 32];
  long long base = ((long long)(b * TT + t)) * (3 * DD) + h * HD + d0;
  float q0 = qkv[base], q1 = qkv[base + 32];
  qkv[base]      = q0 * c0 - q1 * s0;           // rotate_half: [-x2, x1]
  qkv[base + 32] = q1 * c1 + q0 * s1;
  long long kb = base + DD;
  float k0 = qkv[kb], k1 = qkv[kb + 32];
  qkv[kb]      = k0 * c0 - k1 * s0;
  qkv[kb + 32] = k1 * c1 + k0 * s1;
}

// ---------------------------------------------------------------------------
// Causal attention, streaming softmax. One wave per (b,h,query), 4 waves/block.
// out[b,t, h*HD+d]  (matches transpose(0,2,1,3).reshape)
// ---------------------------------------------------------------------------
__global__ __launch_bounds__(128) void attn_kernel(const float* __restrict__ qkv,
                                                   float* __restrict__ out)
{
  const int t = blockIdx.x * 4 + threadIdx.y;
  const int h = blockIdx.y, b = blockIdx.z;
  const int lane = threadIdx.x;
  __shared__ float qs[4][HD];
  long long qbase = ((long long)(b * TT + t)) * (3 * DD) + h * HD;
  qs[threadIdx.y][lane]      = qkv[qbase + lane];
  qs[threadIdx.y][lane + 32] = qkv[qbase + lane + 32];
  __syncthreads();
  const float* q = qs[threadIdx.y];

  const float scale = 0.125f;   // 1/sqrt(64)
  float m = -1e30f, l = 0.f, a0 = 0.f, a1 = 0.f;
  const int d0 = lane, d1 = lane + 32;
  for (int kc = 0; kc <= t; kc += 32) {
    int key = kc + lane;
    float s = -1e30f;
    if (key <= t) {
      const float* kp = qkv + ((long long)(b * TT + key)) * (3 * DD) + DD + h * HD;
      float d = 0.f;
#pragma unroll 8
      for (int j = 0; j < HD; ++j) d += q[j] * kp[j];
      s = d * scale;
    }
    float cm = s;
#pragma unroll
    for (int o = 16; o > 0; o >>= 1) cm = fmaxf(cm, __shfl_xor(cm, o, 32));
    float nm = fmaxf(m, cm);
    float corr = __expf(m - nm);
    a0 *= corr; a1 *= corr; l *= corr;
    float p = (key <= t) ? __expf(s - nm) : 0.f;
    float ps = p;
#pragma unroll
    for (int o = 16; o > 0; o >>= 1) ps += __shfl_xor(ps, o, 32);
    l += ps;
    int lim = min(32, t - kc + 1);
    for (int j = 0; j < lim; ++j) {
      float pj = __shfl(p, j, 32);
      const float* vp = qkv + ((long long)(b * TT + kc + j)) * (3 * DD) + 2 * DD + h * HD;
      a0 += pj * vp[d0];
      a1 += pj * vp[d1];
    }
    m = nm;
  }
  float inv = 1.0f / l;
  long long ob = ((long long)(b * TT + t)) * DD + h * HD;
  out[ob + d0] = a0 * inv;
  out[ob + d1] = a1 * inv;
}

// ---------------------------------------------------------------------------
// Small elementwise kernels
// ---------------------------------------------------------------------------
__global__ void fill_zero_kernel(float* p, long long n) {
  long long i = blockIdx.x * (long long)blockDim.x + threadIdx.x;
  if (i < n) p[i] = 0.f;
}
// o = g * (v - vr)
__global__ void vdyn_kernel(const float* __restrict__ g, const float* __restrict__ v,
                            const float* __restrict__ vr, float* __restrict__ o, long long n) {
  long long i = blockIdx.x * (long long)blockDim.x + threadIdx.x;
  if (i < n) o[i] = g[i] * (v[i] - vr[i]);
}
// S = S * tril ; acc += S      (layout [B,T,T])
__global__ void causal_acc_kernel(float* __restrict__ S, float* __restrict__ acc, long long n) {
  long long i = blockIdx.x * (long long)blockDim.x + threadIdx.x;
  if (i >= n) return;
  int tk = (int)(i % TT);
  int tq = (int)((i / TT) % TT);
  float v = (tk <= tq) ? S[i] : 0.f;
  S[i] = v;
  acc[i] += v;
}
// acc = (acc + S2) * invs * tril
__global__ void osa_tail_kernel(float* __restrict__ acc, const float* __restrict__ S2,
                                float invs, long long n) {
  long long i = blockIdx.x * (long long)blockDim.x + threadIdx.x;
  if (i >= n) return;
  int tk = (int)(i % TT);
  int tq = (int)((i / TT) % TT);
  acc[i] = (tk <= tq) ? (acc[i] + S2[i]) * invs : 0.f;
}
// x += gate * ctx
__global__ void gated_add_kernel(float* __restrict__ x, const float* __restrict__ gate,
                                 const float* __restrict__ ctx, long long n) {
  long long i = blockIdx.x * (long long)blockDim.x + threadIdx.x;
  if (i < n) x[i] += gate[i] * ctx[i];
}

// ---------------------------------------------------------------------------
// Host orchestration
// ---------------------------------------------------------------------------
struct LayerP {
  const float *att_b, *att_g, *f1b, *f1w, *f2b, *f2w;
  const float *mem, *mn_b, *mn_g;
  const float *n_b, *n_g, *opj_b, *opj_w, *pj_b, *pj_w, *qkvw;
  const float *wg_b, *wg_w, *wk_b, *wk_w, *wo_b, *wo_w, *wq_b, *wq_w, *wv_b, *wv_w;
  bool is_mem;
};

static void gemm(hipStream_t st, const float* A, const float* B, const float* bias,
                 float* C, int M, int N, int K,
                 long long sA, long long sB, long long sC, int batches,
                 int transB, int act, float alpha, float beta)
{
  dim3 grid((N + 63) / 64, (M + 63) / 64, batches);
  dim3 blk(32, 4, 1);
  gemm_f16_wmma<<<grid, blk, 0, st>>>(A, B, bias, C, M, N, K, sA, sB, sC,
                                      transB, act, alpha, beta);
}
static void ln(hipStream_t st, const float* X, const float* g, const float* b,
               float* Y, int M)
{
  dim3 grid((M + 7) / 8), blk(32, 8);
  layernorm_kernel<<<grid, blk, 0, st>>>(X, g, b, Y, M);
}
static void ew_grid(long long n, dim3& g, dim3& b) { b = dim3(256); g = dim3((unsigned)((n + 255) / 256)); }

extern "C" void kernel_launch(void* const* d_in, const int* in_sizes, int n_in,
                              void* d_out, int out_size, void* d_ws, size_t ws_size,
                              hipStream_t stream)
{
  const float* x_in = (const float*)d_in[0];
  const float* cosT = (const float*)d_in[1];
  const float* sinT = (const float*)d_in[2];

  // ---- params: flattened pytree (sorted dict keys, lists in order) ----
  int pi = 3;
  auto P = [&]() -> const float* { return (const float*)d_in[pi++]; };

  const float *bf1b = P(), *bf1w = P();            // ffn1 (768->3072)
  const float *bf2b = P(), *bf2w = P();            // ffn2 (3072->768)
  const float *gate_w = P();                       // gate_w
  static const bool NEED_MEM[8] = {false, true, true, false, true, false, true, false};
  LayerP L[8];
  for (int i = 0; i < 8; ++i) {
    LayerP& p = L[i]; p.is_mem = NEED_MEM[i];
    p.att_b = P(); p.att_g = P();
    p.f1b = P(); p.f1w = P();
    p.f2b = P(); p.f2w = P();
    if (p.is_mem) { p.mem = P(); p.mn_b = P(); p.mn_g = P(); }
    p.n_b = P(); p.n_g = P();
    if (p.is_mem) { p.opj_b = P(); p.opj_w = P(); }
    p.pj_b = P(); p.pj_w = P();
    p.qkvw = P();
    if (p.is_mem) {
      p.wg_b = P(); p.wg_w = P();
      p.wk_b = P(); p.wk_w = P();
      p.wo_b = P(); p.wo_w = P();
      p.wq_b = P(); p.wq_w = P();
      p.wv_b = P(); p.wv_w = P();
    }
  }
  const float *nm_b = P(), *nm_g = P();            // norm
  const float *on_b = P(), *on_g = P();            // osa_norm
  const float *osa_proj = P();                     // osa_proj
  const float *vn_b = P(), *vn_g = P();            // v_norm
  const float *wvb_b = P(), *wvb_w = P();          // block w_v

  // ---- workspace carve (floats); total ~111 MB ----
  float* W = (float*)d_ws;
  size_t o = 0;
  float* X   = W + o; o += (size_t)MTOK * DD;
  float* Abuf= W + o; o += (size_t)MTOK * 3072;
  float* QKV = W + o; o += (size_t)MTOK * 2304;
  float* T0  = W + o; o += (size_t)MTOK * DD;
  float* T1  = W + o; o += (size_t)MTOK * DD;
  float* T2  = W + o; o += (size_t)MTOK * DD;
  float* T3  = W + o; o += (size_t)MTOK * DD;
  float* T4  = W + o; o += (size_t)MTOK * DD;
  float* T5  = W + o; o += (size_t)MTOK * DD;
  float* ATT = W + o; o += (size_t)MTOK * DD;
  float* S   = W + o; o += (size_t)BB * TT * TT;
  float* ACC = W + o; o += (size_t)BB * TT * TT;

  const float RS = 1.0f / sqrtf((float)DD);        // 1/sqrt(768)
  const long long NTOKD = (long long)MTOK * DD;
  const long long NSCORE = (long long)BB * TT * TT;
  dim3 eg, eb;

  hipMemcpyAsync(X, x_in, NTOKD * sizeof(float), hipMemcpyDeviceToDevice, stream);
  ew_grid(NSCORE, eg, eb);
  fill_zero_kernel<<<eg, eb, 0, stream>>>(ACC, NSCORE);

  for (int i = 0; i < 8; ++i) {
    const LayerP& p = L[i];
    // x_ffn = x + ffn2(silu(ffn1(ln(x))))
    ln(stream, X, p.n_g, p.n_b, T0, MTOK);
    gemm(stream, T0, p.f1w, p.f1b, Abuf, MTOK, 1152, DD, 0, 0, 0, 1, 0, 1, 1.f, 0.f);
    gemm(stream, Abuf, p.f2w, p.f2b, X, MTOK, DD, 1152, 0, 0, 0, 1, 0, 0, 1.f, 1.f);
    // attn_ctx = proj(attn(rope(qkv(ln(x_ffn)))))
    ln(stream, X, p.att_g, p.att_b, T0, MTOK);
    gemm(stream, T0, p.qkvw, nullptr, QKV, MTOK, 3 * DD, DD, 0, 0, 0, 1, 0, 0, 1.f, 0.f);
    {
      long long n = (long long)BB * TT * NH * (HD / 2);
      ew_grid(n, eg, eb);
      rope_kernel<<<eg, eb, 0, stream>>>(QKV, cosT, sinT);
    }
    attn_kernel<<<dim3(TT / 4, NH, BB), dim3(32, 4), 0, stream>>>(QKV, ATT);
    gemm(stream, ATT, p.pj_w, p.pj_b, X, MTOK, DD, DD, 0, 0, 0, 1, 0, 0, 1.f, 1.f);
    if (!p.is_mem) continue;
    // ---- memory sub-layer ----
    ln(stream, X, p.mn_g, p.mn_b, T0, MTOK);                                   // mc
    gemm(stream, T0, p.wq_w, p.wq_b, T1, MTOK, DD, DD, 0, 0, 0, 1, 0, 1, 1.f, 0.f); // q=silu
    gemm(stream, T0, p.wk_w, p.wk_b, T2, MTOK, DD, DD, 0, 0, 0, 1, 0, 0, 1.f, 0.f); // k
    rownorm_kernel<<<dim3(MTOK / 8), dim3(32, 8), 0, stream>>>(T2, MTOK);
    gemm(stream, T0, p.wv_w, p.wv_b, T3, MTOK, DD, DD, 0, 0, 0, 1, 0, 0, 1.f, 0.f); // v
    gemm(stream, T0, p.wg_w, p.wg_b, T4, MTOK, DD, DD, 0, 0, 0, 1, 0, 2, 1.f, 0.f); // g=sigmoid
    gemm(stream, T2, p.mem, nullptr, T5, MTOK, DD, DD, 0, 0, 0, 1, 0, 0, RS, 0.f);  // v_ret
    ew_grid(NTOKD, eg, eb);
    vdyn_kernel<<<eg, eb, 0, stream>>>(T4, T3, T5, T3, NTOKD);                 // v_dyn
    gemm(stream, T1, p.mem, nullptr, T5, MTOK, DD, DD, 0, 0, 0, 1, 0, 0, RS, 0.f);  // P/sqrt
    gemm(stream, T1, T2, nullptr, S, TT, TT, DD,
         (long long)TT * DD, (long long)TT * DD, (long long)TT * TT, BB,
         1, 0, 1.f, 0.f);                                                      // scores=q@k^T
    ew_grid(NSCORE, eg, eb);
    causal_acc_kernel<<<eg, eb, 0, stream>>>(S, ACC, NSCORE);                  // mask + acc+=
    gemm(stream, S, T3, nullptr, T5, TT, DD, TT,
         (long long)TT * TT, (long long)TT * DD, (long long)TT * DD, BB,
         0, 0, RS, 1.f);                                                       // mem_out
    gemm(stream, T5, p.wo_w, p.wo_b, Abuf, MTOK, 3072, DD, 0, 0, 0, 1, 0, 1, 1.f, 0.f);
    gemm(stream, Abuf, p.opj_w, p.opj_b, X, MTOK, DD, 3072, 0, 0, 0, 1, 0, 0, 1.f, 1.f);
  }

  // ---- block tail ----
  gemm(stream, ACC, X, nullptr, T0, TT, DD, TT,
       (long long)TT * TT, (long long)TT * DD, (long long)TT * DD, BB,
       0, 0, 1.f, 0.f);                                                        // t = acc@x
  gemm(stream, T0, osa_proj, nullptr, T1, MTOK, DD, DD, 0, 0, 0, 1, 0, 1, RS, 0.f); // silu(t@osa/sqrt d)
  ln(stream, T1, on_g, on_b, T2, MTOK);                                        // tn
  gemm(stream, T2, T2, nullptr, S, TT, TT, DD,
       (long long)TT * DD, (long long)TT * DD, (long long)TT * TT, BB,
       1, 0, 1.f, 0.f);                                                        // tn@tn^T
  ew_grid(NSCORE, eg, eb);
  osa_tail_kernel<<<eg, eb, 0, stream>>>(ACC, S, RS, NSCORE);                  // acc final
  gemm(stream, X, wvb_w, wvb_b, T0, MTOK, DD, DD, 0, 0, 0, 1, 0, 0, 1.f, 0.f);
  ln(stream, T0, vn_g, vn_b, T1, MTOK);                                        // v
  gemm(stream, ACC, T1, nullptr, T2, TT, DD, TT,
       (long long)TT * TT, (long long)TT * DD, (long long)TT * DD, BB,
       0, 0, RS, 0.f);                                                         // ctx
  gemm(stream, X, gate_w, nullptr, T3, MTOK, DD, DD, 0, 0, 0, 1, 0, 2, 1.f, 0.f);   // gate
  ew_grid(NTOKD, eg, eb);
  gated_add_kernel<<<eg, eb, 0, stream>>>(X, T3, T2, NTOKD);                   // x += g*ctx
  // final FFN with residual written straight into d_out
  ln(stream, X, nm_g, nm_b, T0, MTOK);
  gemm(stream, T0, bf1w, bf1b, Abuf, MTOK, 3072, DD, 0, 0, 0, 1, 0, 1, 1.f, 0.f);
  hipMemcpyAsync(d_out, X, NTOKD * sizeof(float), hipMemcpyDeviceToDevice, stream);
  gemm(stream, Abuf, bf2w, bf2b, (float*)d_out, MTOK, DD, 3072, 0, 0, 0, 1, 0, 0, 1.f, 1.f);
}
